// StandardAttention_72524817760464
// MI455X (gfx1250) — matrix operand order repaired and checked by reading, not compiled, not run on hardware
//
#include <hip/hip_runtime.h>

#define HIDDEN 1024
#define HEADS 16
#define HEAD_DIM 64
#define BATCH 2
#define SEQ 2048
#define ROWS (BATCH * SEQ) /* 4096 */

typedef __attribute__((ext_vector_type(16))) __bf16 v16bf;
typedef __attribute__((ext_vector_type(8))) float v8f;
typedef __attribute__((ext_vector_type(4))) unsigned int u32x4;
typedef __attribute__((ext_vector_type(8))) int i32x8;
typedef __attribute__((ext_vector_type(4))) int i32x4;
typedef unsigned short u16t;
typedef unsigned int u32t;

union ABFrag {
    uint4 u[2];
    v16bf v;
};
union Acc {
    v8f v;
    float f[8];
};

__device__ __forceinline__ u16t f2bf(float f) {
    u32t u = __float_as_uint(f);
    return (u16t)((u + 0x7FFFu + ((u >> 16) & 1u)) >> 16);
}

__device__ __forceinline__ v8f wmma_bf16(v16bf a, v16bf b, v8f c) {
    // D = A(16x32 bf16) * B(32x16 bf16) + C(16x16 f32)
    return __builtin_amdgcn_wmma_f32_16x16x32_bf16(
        /*neg_a=*/false, a, /*neg_b=*/false, b,
        /*c_mod=*/(short)0, c, /*reuse_a=*/false, /*reuse_b=*/false);
}

// A-matrix fragment (16xK row-major): lane<16 -> row=lane, K = k0+{0..7,16..23}
// lane>=16 -> row=lane-16, K = k0+{8..15,24..31}  (ISA 7.12.2 16-bit A layout)
__device__ __forceinline__ v16bf load_a(const u16t* A, int ld, int m0, int k0) {
    int lane = threadIdx.x & 31;
    int half = lane >> 4, lr = lane & 15;
    const u16t* p = A + (m0 + lr) * ld + k0 + half * 8;
    ABFrag r;
    r.u[0] = *(const uint4*)(p);
    r.u[1] = *(const uint4*)(p + 16);
    return r.v;
}

// B-matrix fragment (KxN, stored as N rows of K contiguous, Bm[n][k]):
// lane<16 -> col n=lane, K = k0+{0..7,16..23} ; lane>=16 -> col n=lane-16, K = k0+{8..15,24..31}
__device__ __forceinline__ v16bf load_b(const u16t* Bm, int ld, int n0, int k0) {
    int lane = threadIdx.x & 31;
    int half = lane >> 4, lr = lane & 15;
    const u16t* p = Bm + (n0 + lr) * ld + k0 + half * 8;
    ABFrag r;
    r.u[0] = *(const uint4*)(p);
    r.u[1] = *(const uint4*)(p + 16);
    return r.v;
}

// ------------------------------------------------------- TDM: 2D tile -> LDS
// Builds the Tensor DMA Descriptor (D#) per CDNA5 ISA and issues
// TENSOR_LOAD_TO_LDS. data_size=2 bytes. Tile stored packed row-major in LDS
// (pitch = tile_k elements). Call from ONE wave only (EXEC ignored by TDM).
__device__ __forceinline__ void tdm_load_2d(unsigned lds_off, const u16t* gsrc,
                                            unsigned tile_k, unsigned tile_rows,
                                            unsigned tensor_k, unsigned tensor_rows,
                                            unsigned row_stride) {
    unsigned long long ga = (unsigned long long)(uintptr_t)gsrc;
    u32x4 g0;
    g0[0] = 1u;                                   // count=1, user mode, no gather
    g0[1] = lds_off;                              // lds_addr (bytes)
    g0[2] = (unsigned)(ga & 0xFFFFFFFFu);         // global_addr[31:0]
    g0[3] = (unsigned)((ga >> 32) & 0x1FFFFFFu)   // global_addr[56:32]
            | (2u << 30);                         // type = 2 ("image")
    i32x8 g1;
    g1[0] = (int)(1u << 16);                      // data_size=1 -> 2 bytes; mask=0
    g1[1] = (int)((tensor_k & 0xFFFFu) << 16);    // tensor_dim0[15:0]
    g1[2] = (int)(((tensor_k >> 16) & 0xFFFFu) | ((tensor_rows & 0xFFFFu) << 16));
    g1[3] = (int)(((tensor_rows >> 16) & 0xFFFFu) | ((tile_k & 0xFFFFu) << 16)); // tile_dim0
    g1[4] = (int)(tile_rows & 0xFFFFu);           // tile_dim1 (tile_dim2 = 0)
    g1[5] = (int)row_stride;                      // tensor_dim0_stride[31:0]
    g1[6] = 0;                                    // stride hi / dim1_stride lo
    g1[7] = 0;
    i32x4 z4 = {0, 0, 0, 0};
#if __has_builtin(__builtin_amdgcn_tensor_load_to_lds)
#if defined(__clang_major__) && (__clang_major__ >= 23)
    i32x8 z8 = {0, 0, 0, 0, 0, 0, 0, 0};
    __builtin_amdgcn_tensor_load_to_lds(g0, g1, z4, z4, z8, 0);
#else
    __builtin_amdgcn_tensor_load_to_lds(g0, g1, z4, z4, 0);
#endif
#else
    (void)g0; (void)g1; (void)z4;
#endif
}

__device__ __forceinline__ void tdm_wait() {
#if __has_builtin(__builtin_amdgcn_s_wait_tensorcnt)
    __builtin_amdgcn_s_wait_tensorcnt(0);
#endif
}

#if __has_builtin(__builtin_amdgcn_tensor_load_to_lds)
#define USE_TDM 1
#else
#define USE_TDM 0
#endif

// fallback cooperative copy (rows x tile_k bf16, packed) if no TDM builtin
__device__ __forceinline__ void copy_tile(u16t* dst, const u16t* src, int rows, int tile_k8,
                                          int ld) {
    for (int idx = threadIdx.x; idx < rows * tile_k8; idx += blockDim.x) {
        int r = idx / tile_k8, c = idx % tile_k8;
        *(uint4*)(dst + (r * tile_k8 + c) * 8) = *(const uint4*)(src + r * ld + c * 8);
    }
}

// ---------------------------------------------------------------- cast f32->bf16
__global__ __launch_bounds__(256) void cast_f32_bf16(const float* __restrict__ in,
                                                     u16t* __restrict__ out, int n8) {
    int i = blockIdx.x * blockDim.x + threadIdx.x;
    if (i < n8) {
        const float4* p = (const float4*)in + i * 2;
        float4 a = p[0], b = p[1];
        union { u16t h[8]; uint4 u; } r;
        r.h[0] = f2bf(a.x); r.h[1] = f2bf(a.y); r.h[2] = f2bf(a.z); r.h[3] = f2bf(a.w);
        r.h[4] = f2bf(b.x); r.h[5] = f2bf(b.y); r.h[6] = f2bf(b.z); r.h[7] = f2bf(b.w);
        ((uint4*)out)[i] = r.u;
    }
}

// ---------------------------------------------------------------- fused QKV GEMM
// block = 256 threads = 8 waves (2x4); block tile 128x256; wave tile 64x64.
// K sliced by 32; double-buffered LDS slices filled by the TDM, one barrier/slice.
__global__ __launch_bounds__(256) void qkv_gemm(
    const u16t* __restrict__ X, const u16t* __restrict__ Wq, const u16t* __restrict__ Wk,
    const u16t* __restrict__ Wv, const float* __restrict__ bq, const float* __restrict__ bk,
    const float* __restrict__ bv, u16t* __restrict__ Q, u16t* __restrict__ K,
    u16t* __restrict__ Vt) {
    __shared__ alignas(16) u16t As[2][128 * 32];  // 2 x  8 KiB
    __shared__ alignas(16) u16t Bs[2][256 * 32];  // 2 x 16 KiB

    int z = blockIdx.z;
    const u16t* W = (z == 0) ? Wq : ((z == 1) ? Wk : Wv);
    const float* bias = (z == 0) ? bq : ((z == 1) ? bk : bv);

    int wave = threadIdx.x >> 5;
    int wr = wave >> 2, wc = wave & 3;
    int Mblk = blockIdx.x * 128, Nblk = blockIdx.y * 256;
    int M0w = wr * 64, N0w = wc * 64;
    const u16t* Ag = X + Mblk * HIDDEN;
    const u16t* Bg = W + Nblk * HIDDEN;

    Acc c[4][4];
#pragma unroll
    for (int mi = 0; mi < 4; ++mi)
#pragma unroll
        for (int ni = 0; ni < 4; ++ni)
#pragma unroll
            for (int i = 0; i < 8; ++i) c[mi][ni].f[i] = 0.f;

    const int NK = HIDDEN / 32;
#if USE_TDM
    if (threadIdx.x < 32) {  // prologue: fill buffer 0
        tdm_load_2d((unsigned)(uintptr_t)&As[0][0], Ag, 32, 128, HIDDEN, ROWS, HIDDEN);
        tdm_load_2d((unsigned)(uintptr_t)&Bs[0][0], Bg, 32, 256, HIDDEN, HIDDEN, HIDDEN);
    }
#endif
    for (int it = 0; it < NK; ++it) {
        int cur = it & 1;
#if USE_TDM
        if (threadIdx.x < 32) tdm_wait();  // slice `it` resident
        __syncthreads();                   // ...and other buffer free
        if (threadIdx.x < 32 && it + 1 < NK) {  // prefetch slice it+1 (overlaps compute)
            int k1 = (it + 1) * 32;
            tdm_load_2d((unsigned)(uintptr_t)&As[1 - cur][0], Ag + k1, 32, 128, HIDDEN, ROWS,
                        HIDDEN);
            tdm_load_2d((unsigned)(uintptr_t)&Bs[1 - cur][0], Bg + k1, 32, 256, HIDDEN, HIDDEN,
                        HIDDEN);
        }
#else
        cur = 0;
        __syncthreads();
        copy_tile(As[0], Ag + it * 32, 128, 4, HIDDEN);
        copy_tile(Bs[0], Bg + it * 32, 256, 4, HIDDEN);
        __syncthreads();
#endif
        v16bf a[4], b[4];
#pragma unroll
        for (int mi = 0; mi < 4; ++mi) a[mi] = load_a(As[cur], 32, M0w + mi * 16, 0);
#pragma unroll
        for (int ni = 0; ni < 4; ++ni) b[ni] = load_b(Bs[cur], 32, N0w + ni * 16, 0);
#pragma unroll
        for (int mi = 0; mi < 4; ++mi)
#pragma unroll
            for (int ni = 0; ni < 4; ++ni) c[mi][ni].v = wmma_bf16(a[mi], b[ni], c[mi][ni].v);
    }

    int lane = threadIdx.x & 31, half = lane >> 4, lr = lane & 15;
#pragma unroll
    for (int mi = 0; mi < 4; ++mi)
#pragma unroll
        for (int ni = 0; ni < 4; ++ni) {
            int col = Nblk + N0w + ni * 16 + lr;
            float bb = bias[col];
            int h = col >> 6, d = col & 63;
#pragma unroll
            for (int i = 0; i < 8; ++i) {
                int gm = Mblk + M0w + mi * 16 + i + half * 8;
                int b_ = gm >> 11, s = gm & (SEQ - 1);
                u16t v = f2bf(c[mi][ni].f[i] + bb);
                if (z == 0)
                    Q[((b_ * HEADS + h) * SEQ + s) * HEAD_DIM + d] = v;
                else if (z == 1)
                    K[((b_ * HEADS + h) * SEQ + s) * HEAD_DIM + d] = v;
                else
                    Vt[((b_ * HEADS + h) * HEAD_DIM + d) * SEQ + s] = v;
            }
        }
}

// ---------------------------------------------------------------- flash attention
// block = 128 threads = 4 waves; each wave owns a 16-row q tile (64 q rows/block).
// K and V^T 64x64 tiles are TDM-staged into double-buffered LDS and shared by all
// 4 waves; online softmax in exp2 domain; P transposed via per-wave LDS scratch.
__global__ __launch_bounds__(128) void attn_kernel(const u16t* __restrict__ Q,
                                                   const u16t* __restrict__ K,
                                                   const u16t* __restrict__ Vt,
                                                   u16t* __restrict__ Out) {
    __shared__ alignas(16) u16t Ks[2][64 * 64];   // 2 x 8 KiB [key][d]
    __shared__ alignas(16) u16t Vs[2][64 * 64];   // 2 x 8 KiB [d][key]
    __shared__ alignas(16) u16t pls[4][16 * 72 + 8];  // per-wave P tile, pitch 72

    int bh = blockIdx.y;
    int wave = threadIdx.x >> 5;
    int q0 = blockIdx.x * 64 + wave * 16;
    const u16t* Qp = Q + (bh * SEQ + q0) * HEAD_DIM;
    const u16t* Kp = K + bh * SEQ * HEAD_DIM;
    const u16t* Vp = Vt + bh * HEAD_DIM * SEQ;
    u16t* pw = &pls[wave][0];
    int lane = threadIdx.x & 31, half = lane >> 4, lr = lane & 15;

    v16bf aq0 = load_a(Qp, HEAD_DIM, 0, 0);
    v16bf aq1 = load_a(Qp, HEAD_DIM, 0, 32);

    Acc o[4];
    float m[8], l[8];
#pragma unroll
    for (int t = 0; t < 4; ++t)
#pragma unroll
        for (int i = 0; i < 8; ++i) o[t].f[i] = 0.f;
#pragma unroll
    for (int i = 0; i < 8; ++i) { m[i] = -1e30f; l[i] = 0.f; }

    const float csc = 0.125f * 1.4426950408889634f;  // (1/sqrt(64)) * log2(e)
    const int NB = SEQ / 64;

#if USE_TDM
    if (threadIdx.x < 32) {  // prologue: key block 0
        tdm_load_2d((unsigned)(uintptr_t)&Ks[0][0], Kp, 64, 64, HEAD_DIM, SEQ, HEAD_DIM);
        tdm_load_2d((unsigned)(uintptr_t)&Vs[0][0], Vp, 64, 64, SEQ, HEAD_DIM, SEQ);
    }
#endif
    for (int it = 0; it < NB; ++it) {
        int cur = it & 1;
#if USE_TDM
        if (threadIdx.x < 32) tdm_wait();
        __syncthreads();
        if (threadIdx.x < 32 && it + 1 < NB) {
            int kb1 = (it + 1) * 64;
            tdm_load_2d((unsigned)(uintptr_t)&Ks[1 - cur][0], Kp + kb1 * HEAD_DIM, 64, 64,
                        HEAD_DIM, SEQ, HEAD_DIM);
            tdm_load_2d((unsigned)(uintptr_t)&Vs[1 - cur][0], Vp + kb1, 64, 64, SEQ, HEAD_DIM,
                        SEQ);
        }
#else
        cur = 0;
        __syncthreads();
        copy_tile(Ks[0], Kp + it * 64 * HEAD_DIM, 64, 8, HEAD_DIM);
        copy_tile(Vs[0], Vp + it * 64, 64, 8, SEQ);
        __syncthreads();
#endif
        Acc s[4];
#pragma unroll
        for (int j = 0; j < 4; ++j) {
            v16bf bk0 = load_b(Ks[cur], 64, j * 16, 0);
            v16bf bk1 = load_b(Ks[cur], 64, j * 16, 32);
            v8f zz = {0.f, 0.f, 0.f, 0.f, 0.f, 0.f, 0.f, 0.f};
            zz = wmma_bf16(aq0, bk0, zz);
            zz = wmma_bf16(aq1, bk1, zz);
            s[j].v = zz;
        }
        float mn[8];
#pragma unroll
        for (int i = 0; i < 8; ++i) {
            float mx = -1e30f;
#pragma unroll
            for (int j = 0; j < 4; ++j) {
                float t = s[j].f[i] * csc;
                s[j].f[i] = t;
                mx = fmaxf(mx, t);
            }
            mn[i] = mx;
        }
#pragma unroll
        for (int msk = 1; msk <= 8; msk <<= 1)
#pragma unroll
            for (int i = 0; i < 8; ++i) mn[i] = fmaxf(mn[i], __shfl_xor(mn[i], msk, 32));

        float alpha[8], rs[8];
#pragma unroll
        for (int i = 0; i < 8; ++i) {
            float mu = fmaxf(m[i], mn[i]);
            alpha[i] = exp2f(m[i] - mu);
            m[i] = mu;
            rs[i] = 0.f;
        }
#pragma unroll
        for (int j = 0; j < 4; ++j)
#pragma unroll
            for (int i = 0; i < 8; ++i) {
                float p = exp2f(s[j].f[i] - m[i]);
                s[j].f[i] = p;
                rs[i] += p;
            }
#pragma unroll
        for (int msk = 1; msk <= 8; msk <<= 1)
#pragma unroll
            for (int i = 0; i < 8; ++i) rs[i] += __shfl_xor(rs[i], msk, 32);
#pragma unroll
        for (int i = 0; i < 8; ++i) l[i] = l[i] * alpha[i] + rs[i];
#pragma unroll
        for (int t = 0; t < 4; ++t)
#pragma unroll
            for (int i = 0; i < 8; ++i) o[t].f[i] *= alpha[i];

        // transpose P (C layout -> A layout) through per-wave LDS scratch
#pragma unroll
        for (int j = 0; j < 4; ++j)
#pragma unroll
            for (int i = 0; i < 8; ++i) {
                int row = i + half * 8, col = j * 16 + lr;
                pw[row * 72 + col] = f2bf(s[j].f[i]);
            }
        v16bf ap0 = load_a(pw, 72, 0, 0);
        v16bf ap1 = load_a(pw, 72, 0, 32);
#pragma unroll
        for (int t = 0; t < 4; ++t) {
            v16bf bv0 = load_b(Vs[cur], 64, t * 16, 0);
            v16bf bv1 = load_b(Vs[cur], 64, t * 16, 32);
            o[t].v = wmma_bf16(ap0, bv0, o[t].v);
            o[t].v = wmma_bf16(ap1, bv1, o[t].v);
        }
    }

    int b = bh >> 4, h = bh & 15;
#pragma unroll
    for (int t = 0; t < 4; ++t)
#pragma unroll
        for (int i = 0; i < 8; ++i) {
            int row = q0 + i + half * 8;
            int gm = b * SEQ + row;
            int col = h * HEAD_DIM + t * 16 + lr;
            Out[gm * HIDDEN + col] = f2bf(o[t].f[i] / l[i]);
        }
}

// ---------------------------------------------------------------- output projection
__global__ __launch_bounds__(256) void out_gemm(const u16t* __restrict__ A,
                                                const u16t* __restrict__ Wo,
                                                const float* __restrict__ bo,
                                                float* __restrict__ Out) {
    __shared__ alignas(16) u16t As[2][128 * 32];
    __shared__ alignas(16) u16t Bs[2][256 * 32];

    int wave = threadIdx.x >> 5;
    int wr = wave >> 2, wc = wave & 3;
    int Mblk = blockIdx.x * 128, Nblk = blockIdx.y * 256;
    int M0w = wr * 64, N0w = wc * 64;
    const u16t* Ag = A + Mblk * HIDDEN;
    const u16t* Bg = Wo + Nblk * HIDDEN;

    Acc c[4][4];
#pragma unroll
    for (int mi = 0; mi < 4; ++mi)
#pragma unroll
        for (int ni = 0; ni < 4; ++ni)
#pragma unroll
            for (int i = 0; i < 8; ++i) c[mi][ni].f[i] = 0.f;

    const int NK = HIDDEN / 32;
#if USE_TDM
    if (threadIdx.x < 32) {
        tdm_load_2d((unsigned)(uintptr_t)&As[0][0], Ag, 32, 128, HIDDEN, ROWS, HIDDEN);
        tdm_load_2d((unsigned)(uintptr_t)&Bs[0][0], Bg, 32, 256, HIDDEN, HIDDEN, HIDDEN);
    }
#endif
    for (int it = 0; it < NK; ++it) {
        int cur = it & 1;
#if USE_TDM
        if (threadIdx.x < 32) tdm_wait();
        __syncthreads();
        if (threadIdx.x < 32 && it + 1 < NK) {
            int k1 = (it + 1) * 32;
            tdm_load_2d((unsigned)(uintptr_t)&As[1 - cur][0], Ag + k1, 32, 128, HIDDEN, ROWS,
                        HIDDEN);
            tdm_load_2d((unsigned)(uintptr_t)&Bs[1 - cur][0], Bg + k1, 32, 256, HIDDEN, HIDDEN,
                        HIDDEN);
        }
#else
        cur = 0;
        __syncthreads();
        copy_tile(As[0], Ag + it * 32, 128, 4, HIDDEN);
        copy_tile(Bs[0], Bg + it * 32, 256, 4, HIDDEN);
        __syncthreads();
#endif
        v16bf a[4], b[4];
#pragma unroll
        for (int mi = 0; mi < 4; ++mi) a[mi] = load_a(As[cur], 32, M0w + mi * 16, 0);
#pragma unroll
        for (int ni = 0; ni < 4; ++ni) b[ni] = load_b(Bs[cur], 32, N0w + ni * 16, 0);
#pragma unroll
        for (int mi = 0; mi < 4; ++mi)
#pragma unroll
            for (int ni = 0; ni < 4; ++ni) c[mi][ni].v = wmma_bf16(a[mi], b[ni], c[mi][ni].v);
    }

    int lane = threadIdx.x & 31, half = lane >> 4, lr = lane & 15;
#pragma unroll
    for (int mi = 0; mi < 4; ++mi)
#pragma unroll
        for (int ni = 0; ni < 4; ++ni) {
            int col = Nblk + N0w + ni * 16 + lr;
            float bb = bo[col];
#pragma unroll
            for (int i = 0; i < 8; ++i) {
                int gm = Mblk + M0w + mi * 16 + i + half * 8;
                Out[gm * HIDDEN + col] = c[mi][ni].f[i] + bb;
            }
        }
}

// ---------------------------------------------------------------- launcher
extern "C" void kernel_launch(void* const* d_in, const int* in_sizes, int n_in, void* d_out,
                              int out_size, void* d_ws, size_t ws_size, hipStream_t stream) {
    const float* x = (const float*)d_in[0];
    const float* Wq = (const float*)d_in[1];
    const float* bq = (const float*)d_in[2];
    const float* Wk = (const float*)d_in[3];
    const float* bk = (const float*)d_in[4];
    const float* Wv = (const float*)d_in[5];
    const float* bv = (const float*)d_in[6];
    const float* Wo = (const float*)d_in[7];
    const float* bo = (const float*)d_in[8];

    char* ws = (char*)d_ws;
    const size_t MB = 1u << 20;
    u16t* xb = (u16t*)(ws + 0 * MB);    //  8 MiB: x bf16 [4096][1024]
    u16t* wqb = (u16t*)(ws + 8 * MB);   //  2 MiB each
    u16t* wkb = (u16t*)(ws + 10 * MB);
    u16t* wvb = (u16t*)(ws + 12 * MB);
    u16t* wob = (u16t*)(ws + 14 * MB);
    u16t* Qb = (u16t*)(ws + 16 * MB);   //  8 MiB [B,H,S,D]
    u16t* Kb = (u16t*)(ws + 24 * MB);   //  8 MiB [B,H,S,D]
    u16t* Vtb = (u16t*)(ws + 32 * MB);  //  8 MiB [B,H,D,S]
    u16t* attn = (u16t*)(ws + 40 * MB); //  8 MiB [4096][1024]

    cast_f32_bf16<<<(ROWS * HIDDEN / 8 + 255) / 256, 256, 0, stream>>>(x, xb, ROWS * HIDDEN / 8);
    int w8 = HIDDEN * HIDDEN / 8;
    cast_f32_bf16<<<(w8 + 255) / 256, 256, 0, stream>>>(Wq, wqb, w8);
    cast_f32_bf16<<<(w8 + 255) / 256, 256, 0, stream>>>(Wk, wkb, w8);
    cast_f32_bf16<<<(w8 + 255) / 256, 256, 0, stream>>>(Wv, wvb, w8);
    cast_f32_bf16<<<(w8 + 255) / 256, 256, 0, stream>>>(Wo, wob, w8);

    qkv_gemm<<<dim3(ROWS / 128, HIDDEN / 256, 3), 256, 0, stream>>>(xb, wqb, wkb, wvb, bq, bk, bv,
                                                                    Qb, Kb, Vtb);
    attn_kernel<<<dim3(SEQ / 64, BATCH * HEADS), 128, 0, stream>>>(Qb, Kb, Vtb, attn);
    out_gemm<<<dim3(ROWS / 128, HIDDEN / 256), 256, 0, stream>>>(attn, wob, bo, (float*)d_out);
}
